// Condition9RotR_24799141167113
// MI455X (gfx1250) — matrix operand order repaired and checked by reading, not compiled, not run on hardware
//
#include <hip/hip_runtime.h>

typedef __attribute__((ext_vector_type(16))) __bf16 v16bf;
typedef __attribute__((ext_vector_type(8)))  __bf16 v8bf;
typedef __attribute__((ext_vector_type(4)))  __bf16 v4bf;
typedef __attribute__((ext_vector_type(8)))  float  v8f;
typedef __attribute__((ext_vector_type(4)))  float  v4f;

#define NPTS 200000
#define FDIM 512
#define HID  256
#define WAVES 4
#define ROWS_PER_BLOCK (WAVES * 16)

#define W1_KB 16   // k-blocks (of 32) in W1
#define W2_KB 8    // k-blocks in W2 / W3

// packed weight sizes in elements: ntiles * kblocks * 32 lanes * 16 elems
#define W1P_ELEMS (16 * W1_KB * 512)
#define W2P_ELEMS (16 * W2_KB * 512)
#define W3P_ELEMS (1 * W2_KB * 512)

__device__ __forceinline__ __bf16 f2bf(float f) { return (__bf16)f; }  // native v_cvt

// Pack W[K x C] (row-major f32) into the WMMA B-matrix lane layout (bf16):
// dst[((nt*kblocks + kb)*32 + lane)*16 + e] = W[kb*32 + (e>>3)*16 + (lane>>4)*8 + (e&7)][nt*16 + (lane&15)]
__global__ void pack_b_kernel(const float* __restrict__ W, __bf16* __restrict__ dst,
                              int K, int C, int kblocks, int total) {
    int idx = blockIdx.x * blockDim.x + threadIdx.x;
    if (idx >= total) return;
    int e    = idx & 15;
    int lane = (idx >> 4) & 31;
    int rest = idx >> 9;
    int kb   = rest % kblocks;
    int nt   = rest / kblocks;
    int k = kb * 32 + (e >> 3) * 16 + (lane >> 4) * 8 + (e & 7);
    int c = nt * 16 + (lane & 15);
    float v = (k < K && c < C) ? W[(size_t)k * C + c] : 0.0f;
    dst[idx] = f2bf(v);
}

// cooperative stage of one 16KB K-slice (16 ntiles x 32 lanes x 16 bf16) via registers
template<int KB>
__device__ __forceinline__ void stage_load(const __bf16* __restrict__ wp, int ks, int tid,
                                           v8bf t[8]) {
#pragma unroll
    for (int i = 0; i < 8; ++i) {
        int u  = tid + i * 128;          // 1024 16-byte units / 128 threads
        int nt = u >> 6;
        int r  = u & 63;
        t[i] = *(const v8bf*)(wp + (((size_t)(nt * KB + ks)) << 9) + (size_t)r * 8);
    }
}
__device__ __forceinline__ void stage_store(__bf16* dst, int tid, const v8bf t[8]) {
#pragma unroll
    for (int i = 0; i < 8; ++i) {
        int u = tid + i * 128;
        *(v8bf*)(dst + (size_t)u * 8) = t[i];
    }
}

__device__ __forceinline__ v16bf load_a_lds(const __bf16* ap) {
    v8bf lo = *(const v8bf*)(ap);        // K = kb + hf*8 + 0..7
    v8bf hi = *(const v8bf*)(ap + 16);   // K = kb + 16 + hf*8 + 0..7
    v16bf a;
#pragma unroll
    for (int i = 0; i < 8; ++i) { a[i] = lo[i]; a[8 + i] = hi[i]; }
    return a;
}

__global__ __launch_bounds__(128, 1)
void mlp_polar_kernel(const float* __restrict__ feature,
                      const float* __restrict__ rotation,
                      const float* __restrict__ b1,
                      const float* __restrict__ b2,
                      const float* __restrict__ b3,
                      const __bf16* __restrict__ w1p,
                      const __bf16* __restrict__ w2p,
                      const __bf16* __restrict__ w3p,
                      float* __restrict__ out) {
    __shared__ __bf16 stg[2][16 * 512];       // 2 x 16KB weight slices (double buffer)
    __shared__ __bf16 hbuf[WAVES][16 * HID];  // per-wave activations (h1, then h2, then f32 tile)

    const int tid  = threadIdx.x;
    const int wave = tid >> 5;
    const int lane = tid & 31;
    const int hf   = lane >> 4;     // half-wave select
    const int col  = lane & 15;     // B/C column, also A row
    const int m0   = blockIdx.x * ROWS_PER_BLOCK + wave * 16;

    __bf16* h = &hbuf[wave][0];

    // ---------------- Layer 1: feature[16x512] @ W1 -> h[16x256] ----------------
    v8f acc[16];
#pragma unroll
    for (int nt = 0; nt < 16; ++nt) acc[nt] = v8f{};

    {
        v8bf t[8];
        stage_load<W1_KB>(w1p, 0, tid, t);
        stage_store(stg[0], tid, t);
    }
    __syncthreads();

    const float* xrow = feature + (size_t)(m0 + col) * FDIM + hf * 8;
    for (int ks = 0; ks < W1_KB; ++ks) {
        const int cur  = ks & 1;
        const bool more = (ks + 1 < W1_KB);
        v8bf t[8];
        if (more) stage_load<W1_KB>(w1p, ks + 1, tid, t);   // issue next slice early

        const float* p = xrow + ks * 32;
        v4f f0 = *(const v4f*)(p);
        v4f f1 = *(const v4f*)(p + 4);
        v4f f2 = *(const v4f*)(p + 16);
        v4f f3 = *(const v4f*)(p + 20);
        v4bf a0 = __builtin_convertvector(f0, v4bf);
        v4bf a1 = __builtin_convertvector(f1, v4bf);
        v4bf a2 = __builtin_convertvector(f2, v4bf);
        v4bf a3 = __builtin_convertvector(f3, v4bf);
        v16bf a;
#pragma unroll
        for (int i = 0; i < 4; ++i) {
            a[i] = a0[i]; a[4 + i] = a1[i]; a[8 + i] = a2[i]; a[12 + i] = a3[i];
        }
#pragma unroll
        for (int nt = 0; nt < 16; ++nt) {
            v16bf b = *(const v16bf*)(&stg[cur][nt * 512 + lane * 16]);
            acc[nt] = __builtin_amdgcn_wmma_f32_16x16x32_bf16(
                false, a, false, b, (short)0, acc[nt], false, false);
        }
        if (more) stage_store(stg[cur ^ 1], tid, t);
        __syncthreads();
    }
#pragma unroll
    for (int nt = 0; nt < 16; ++nt) {
        float bias = b1[nt * 16 + col];
#pragma unroll
        for (int j = 0; j < 8; ++j) {
            float v = acc[nt][j] + bias;
            v = v > 0.0f ? v : 0.0f;
            h[(j + 8 * hf) * HID + nt * 16 + col] = f2bf(v);
        }
    }

    // ---------------- Layer 2: h[16x256] @ W2 -> h[16x256] (in-place via regs) -----
    v8f acc2[16];
#pragma unroll
    for (int nt = 0; nt < 16; ++nt) acc2[nt] = v8f{};

    {
        v8bf t[8];
        stage_load<W2_KB>(w2p, 0, tid, t);
        stage_store(stg[0], tid, t);
    }
    __syncthreads();

    for (int ks = 0; ks < W2_KB; ++ks) {
        const int cur  = ks & 1;
        const bool more = (ks + 1 < W2_KB);
        v8bf t[8];
        if (more) stage_load<W2_KB>(w2p, ks + 1, tid, t);

        v16bf a = load_a_lds(h + col * HID + ks * 32 + hf * 8);
#pragma unroll
        for (int nt = 0; nt < 16; ++nt) {
            v16bf b = *(const v16bf*)(&stg[cur][nt * 512 + lane * 16]);
            acc2[nt] = __builtin_amdgcn_wmma_f32_16x16x32_bf16(
                false, a, false, b, (short)0, acc2[nt], false, false);
        }
        if (more) stage_store(stg[cur ^ 1], tid, t);
        __syncthreads();
    }
    // all reads of h by this wave are complete (same-wave LDS ops retire in order):
    // overwrite h with layer-2 activations
#pragma unroll
    for (int nt = 0; nt < 16; ++nt) {
        float bias = b2[nt * 16 + col];
#pragma unroll
        for (int j = 0; j < 8; ++j) {
            float v = acc2[nt][j] + bias;
            v = v > 0.0f ? v : 0.0f;
            h[(j + 8 * hf) * HID + nt * 16 + col] = f2bf(v);
        }
    }
    __syncthreads();

    // ---------------- Layer 3: h[16x256] @ W3(pad16) -> 16x16 tile ----------------
    v8f acc3 = v8f{};
    for (int ks = 0; ks < W2_KB; ++ks) {
        v16bf a = load_a_lds(h + col * HID + ks * 32 + hf * 8);
        v16bf b = *(const v16bf*)(w3p + ((size_t)ks * 32 + lane) * 16);
        acc3 = __builtin_amdgcn_wmma_f32_16x16x32_bf16(
            false, a, false, b, (short)0, acc3, false, false);
    }
    float* ot = reinterpret_cast<float*>(h);  // reuse as 16x16 f32 tile
#pragma unroll
    for (int j = 0; j < 8; ++j) ot[(j + 8 * hf) * 16 + col] = acc3[j];
    __syncthreads();

    // ---------------- Epilogue: mat=out9+I; X = rotation@mat; polar(X) ----------------
    if (lane < 16) {
        const int g = m0 + lane;
        float m9[9];
#pragma unroll
        for (int i = 0; i < 9; ++i) m9[i] = ot[lane * 16 + i] + b3[i];
        m9[0] += 1.0f; m9[4] += 1.0f; m9[8] += 1.0f;

        const float* R = rotation + (size_t)g * 9;
        float r9[9];
#pragma unroll
        for (int i = 0; i < 9; ++i) r9[i] = R[i];

        float X[9];
#pragma unroll
        for (int a = 0; a < 3; ++a)
#pragma unroll
            for (int b = 0; b < 3; ++b)
                X[3 * a + b] = r9[3 * a + 0] * m9[b] +
                               r9[3 * a + 1] * m9[3 + b] +
                               r9[3 * a + 2] * m9[6 + b];

        // Scaled Newton iteration for the orthogonal polar factor (== U @ Vh)
#pragma unroll
        for (int it = 0; it < 8; ++it) {
            float c0 = X[4] * X[8] - X[5] * X[7];
            float c1 = X[5] * X[6] - X[3] * X[8];
            float c2 = X[3] * X[7] - X[4] * X[6];
            float c3 = X[2] * X[7] - X[1] * X[8];
            float c4 = X[0] * X[8] - X[2] * X[6];
            float c5 = X[1] * X[6] - X[0] * X[7];
            float c6 = X[1] * X[5] - X[2] * X[4];
            float c7 = X[2] * X[3] - X[0] * X[5];
            float c8 = X[0] * X[4] - X[1] * X[3];
            float det = X[0] * c0 + X[1] * c1 + X[2] * c2;
            float guard = det >= 0.0f ? 1e-30f : -1e-30f;
            float invdet = 1.0f / (det + guard);
            float nX = X[0]*X[0]+X[1]*X[1]+X[2]*X[2]+X[3]*X[3]+X[4]*X[4]
                     + X[5]*X[5]+X[6]*X[6]+X[7]*X[7]+X[8]*X[8] + 1e-30f;
            float nC = c0*c0+c1*c1+c2*c2+c3*c3+c4*c4+c5*c5+c6*c6+c7*c7+c8*c8;
            float gam = sqrtf(sqrtf(nC * invdet * invdet / nX + 1e-30f));
            float ga = 0.5f * gam;
            float gb = 0.5f * invdet / gam;
            X[0] = ga * X[0] + gb * c0;  X[1] = ga * X[1] + gb * c1;  X[2] = ga * X[2] + gb * c2;
            X[3] = ga * X[3] + gb * c3;  X[4] = ga * X[4] + gb * c4;  X[5] = ga * X[5] + gb * c5;
            X[6] = ga * X[6] + gb * c6;  X[7] = ga * X[7] + gb * c7;  X[8] = ga * X[8] + gb * c8;
        }

        float* ro = out + (size_t)g * 9;
#pragma unroll
        for (int i = 0; i < 9; ++i) ro[i] = X[i];
        out[(size_t)NPTS * 9 + g] = 0.0f;   // logdet
    }
}

extern "C" void kernel_launch(void* const* d_in, const int* in_sizes, int n_in,
                              void* d_out, int out_size, void* d_ws, size_t ws_size,
                              hipStream_t stream) {
    const float* rotation = (const float*)d_in[0];
    const float* feature  = (const float*)d_in[1];
    const float* W1 = (const float*)d_in[2];
    const float* b1 = (const float*)d_in[3];
    const float* W2 = (const float*)d_in[4];
    const float* b2 = (const float*)d_in[5];
    const float* W3 = (const float*)d_in[6];
    const float* b3 = (const float*)d_in[7];

    char* ws = (char*)d_ws;
    __bf16* w1p = (__bf16*)(ws);
    __bf16* w2p = (__bf16*)(ws + (size_t)W1P_ELEMS * 2);
    __bf16* w3p = (__bf16*)(ws + (size_t)(W1P_ELEMS + W2P_ELEMS) * 2);

    pack_b_kernel<<<(W1P_ELEMS + 255) / 256, 256, 0, stream>>>(W1, w1p, 512, 256, W1_KB, W1P_ELEMS);
    pack_b_kernel<<<(W2P_ELEMS + 255) / 256, 256, 0, stream>>>(W2, w2p, 256, 256, W2_KB, W2P_ELEMS);
    pack_b_kernel<<<(W3P_ELEMS + 255) / 256, 256, 0, stream>>>(W3, w3p, 256, 9,   W2_KB, W3P_ELEMS);

    mlp_polar_kernel<<<NPTS / ROWS_PER_BLOCK, 128, 0, stream>>>(
        feature, rotation, b1, b2, b3, w1p, w2p, w3p, (float*)d_out);
}
